// Model_63256278335531
// MI455X (gfx1250) — matrix operand order repaired and checked by reading, not compiled, not run on hardware
//
#include <hip/hip_runtime.h>

typedef __bf16 bf16_t;
typedef __attribute__((ext_vector_type(16))) __bf16 v16bf;
typedef __attribute__((ext_vector_type(8)))  __bf16 v8bf;
typedef __attribute__((ext_vector_type(4)))  __bf16 v4bf;
typedef __attribute__((ext_vector_type(8)))  float  v8f;

union FragBF { v16bf f; v8bf h[2]; };

#define BLK_M 128
#define BLK_N 128
#define BLK_K 64
#define THREADS 256
#define LDS_STRIDE (BLK_K + 8)   // 72 bf16 = 144B rows: 16B-aligned, breaks bank conflicts

// ---------------------------------------------------------------------------
// Transpose + convert weights: W[k][n] (f32, row-major [K][N]) -> WT[n][k] bf16
// ---------------------------------------------------------------------------
__global__ void transpose_bf16(const float* __restrict__ W, bf16_t* __restrict__ WT,
                               int K, int Nn) {
    int idx = blockIdx.x * blockDim.x + threadIdx.x;
    if (idx >= K * Nn) return;
    int n = idx % Nn;
    int k = idx / Nn;
    WT[(size_t)n * K + k] = (bf16_t)W[(size_t)k * Nn + n];
}

// ---------------------------------------------------------------------------
// Single-B GEMM: C[M x 512] = A[M x K](f32) * BT[512 x K]^T (bf16, N-major)
//  STORE_F32=false: CT[n][m] = (bf16)acc  -> bf16 N-major [512][M]
// (used for layer-2 feature GEMMs where A matrices are distinct)
// ---------------------------------------------------------------------------
template <bool STORE_F32>
__global__ __launch_bounds__(THREADS)
void gemm_bf16_wmma(const float* __restrict__ A, int lda,
                    const bf16_t* __restrict__ BT, int ldb,
                    float* __restrict__ C, int ldc,
                    bf16_t* __restrict__ CT, int ldct,
                    const float* __restrict__ bias,
                    const float* __restrict__ alpha,
                    int K)
{
    __shared__ bf16_t sA[BLK_M][LDS_STRIDE];
    __shared__ bf16_t sB[BLK_N][LDS_STRIDE];

    const int tid   = threadIdx.x;
    const int lane  = tid & 31;
    const int wave  = tid >> 5;
    const int waveM = wave & 1;
    const int waveN = wave >> 1;
    const int row   = lane & 15;
    const int grp   = lane >> 4;

    const int m0 = blockIdx.y * BLK_M;
    const int n0 = blockIdx.x * BLK_N;

    v8f acc[4][2];
#pragma unroll
    for (int mi = 0; mi < 4; ++mi)
#pragma unroll
        for (int ni = 0; ni < 2; ++ni)
            acc[mi][ni] = (v8f){0.f,0.f,0.f,0.f,0.f,0.f,0.f,0.f};

    const int nStages = K / BLK_K;

    float4 aPre[8];
    v8bf   bPre[4];

#pragma unroll
    for (int i = 0; i < 8; ++i) {
        int idx = tid + i * THREADS; int r = idx >> 4, c = idx & 15;
        aPre[i] = *reinterpret_cast<const float4*>(A + (size_t)(m0 + r) * lda + c * 4);
    }
#pragma unroll
    for (int i = 0; i < 4; ++i) {
        int idx = tid + i * THREADS; int r = idx >> 3, c = idx & 7;
        bPre[i] = *reinterpret_cast<const v8bf*>(BT + (size_t)(n0 + r) * ldb + c * 8);
    }
#pragma unroll
    for (int i = 0; i < 8; ++i) {
        int idx = tid + i * THREADS; int r = idx >> 4, c = idx & 15;
        v4bf t; t[0] = (bf16_t)aPre[i].x; t[1] = (bf16_t)aPre[i].y;
                t[2] = (bf16_t)aPre[i].z; t[3] = (bf16_t)aPre[i].w;
        *reinterpret_cast<v4bf*>(&sA[r][c * 4]) = t;
    }
#pragma unroll
    for (int i = 0; i < 4; ++i) {
        int idx = tid + i * THREADS; int r = idx >> 3, c = idx & 7;
        *reinterpret_cast<v8bf*>(&sB[r][c * 8]) = bPre[i];
    }
    __syncthreads();

    for (int s = 0; s < nStages; ++s) {
        const int kNext = (s + 1) * BLK_K;
        if (s + 1 < nStages) {
#pragma unroll
            for (int i = 0; i < 8; ++i) {
                int idx = tid + i * THREADS; int r = idx >> 4, c = idx & 15;
                aPre[i] = *reinterpret_cast<const float4*>(
                    A + (size_t)(m0 + r) * lda + kNext + c * 4);
            }
#pragma unroll
            for (int i = 0; i < 4; ++i) {
                int idx = tid + i * THREADS; int r = idx >> 3, c = idx & 7;
                bPre[i] = *reinterpret_cast<const v8bf*>(
                    BT + (size_t)(n0 + r) * ldb + kNext + c * 8);
            }
        }

#pragma unroll
        for (int kk = 0; kk < BLK_K; kk += 32) {
            FragBF bF[2];
#pragma unroll
            for (int ni = 0; ni < 2; ++ni) {
                int rb = waveN * 32 + ni * 16 + row;
                bF[ni].h[0] = *reinterpret_cast<const v8bf*>(&sB[rb][kk + grp * 16]);
                bF[ni].h[1] = *reinterpret_cast<const v8bf*>(&sB[rb][kk + grp * 16 + 8]);
            }
#pragma unroll
            for (int mi = 0; mi < 4; ++mi) {
                int ra = waveM * 64 + mi * 16 + row;
                FragBF aF;
                aF.h[0] = *reinterpret_cast<const v8bf*>(&sA[ra][kk + grp * 16]);
                aF.h[1] = *reinterpret_cast<const v8bf*>(&sA[ra][kk + grp * 16 + 8]);
#pragma unroll
                for (int ni = 0; ni < 2; ++ni) {
                    acc[mi][ni] = __builtin_amdgcn_wmma_f32_16x16x32_bf16(
                        false, aF.f, false, bF[ni].f, (short)0, acc[mi][ni],
                        false, false);
                }
            }
        }
        __syncthreads();
        if (s + 1 < nStages) {
#pragma unroll
            for (int i = 0; i < 8; ++i) {
                int idx = tid + i * THREADS; int r = idx >> 4, c = idx & 15;
                v4bf t; t[0] = (bf16_t)aPre[i].x; t[1] = (bf16_t)aPre[i].y;
                        t[2] = (bf16_t)aPre[i].z; t[3] = (bf16_t)aPre[i].w;
                *reinterpret_cast<v4bf*>(&sA[r][c * 4]) = t;
            }
#pragma unroll
            for (int i = 0; i < 4; ++i) {
                int idx = tid + i * THREADS; int r = idx >> 3, c = idx & 7;
                *reinterpret_cast<v8bf*>(&sB[r][c * 8]) = bPre[i];
            }
            __syncthreads();
        }
    }

    const float alphav = STORE_F32 ? alpha[0] : 0.f;
#pragma unroll
    for (int mi = 0; mi < 4; ++mi) {
#pragma unroll
        for (int ni = 0; ni < 2; ++ni) {
            const int n  = n0 + waveN * 32 + ni * 16 + row;
            const int mB = m0 + waveM * 64 + mi * 16 + grp * 8;
            if (STORE_F32) {
                const float bv = bias[n];
#pragma unroll
                for (int r = 0; r < 8; ++r) {
                    float v = acc[mi][ni][r] + bv;
                    v = (v >= 0.f) ? v : alphav * v;
                    C[(size_t)(mB + r) * ldc + n] = v;
                }
            } else {
                v8bf t;
#pragma unroll
                for (int r = 0; r < 8; ++r) t[r] = (bf16_t)acc[mi][ni][r];
                *reinterpret_cast<v8bf*>(&CT[(size_t)n * ldct + mB]) = t;
            }
        }
    }
}

// ---------------------------------------------------------------------------
// Dual-B GEMM: one pass over A computes A*B0 and A*B1 (pos/neg share A and
// share bias/alpha) -> halves adj/diff HBM traffic, each A fragment feeds
// 4 WMMAs. B tiles are bf16 in memory already, so they are DMA'd straight
// into a double-buffered LDS region with GLOBAL_LOAD_ASYNC_TO_LDS_B128
// (ASYNCcnt), issued BEFORE the current stage's WMMA block so the copy
// overlaps matrix math and uses zero staging VGPRs.
// ---------------------------------------------------------------------------
template <bool STORE_F32>
__global__ __launch_bounds__(THREADS)
void gemm_dual_bf16_wmma(const float* __restrict__ A, int lda,
                         const bf16_t* __restrict__ BT0,
                         const bf16_t* __restrict__ BT1, int ldb,
                         float* __restrict__ C0, float* __restrict__ C1, int ldc,
                         bf16_t* __restrict__ CT0, bf16_t* __restrict__ CT1, int ldct,
                         const float* __restrict__ bias,
                         const float* __restrict__ alpha,
                         int K)
{
    __shared__ bf16_t sA[BLK_M][LDS_STRIDE];
    __shared__ bf16_t sB[2][2][BLK_N][LDS_STRIDE];   // [buffer][B-operand]

    const int tid   = threadIdx.x;
    const int lane  = tid & 31;
    const int wave  = tid >> 5;
    const int waveM = wave & 1;
    const int waveN = wave >> 1;
    const int row   = lane & 15;
    const int grp   = lane >> 4;

    const int m0 = blockIdx.y * BLK_M;
    const int n0 = blockIdx.x * BLK_N;

    const bf16_t* BT[2] = { BT0, BT1 };

    v8f acc[2][4][2];
#pragma unroll
    for (int b = 0; b < 2; ++b)
#pragma unroll
        for (int mi = 0; mi < 4; ++mi)
#pragma unroll
            for (int ni = 0; ni < 2; ++ni)
                acc[b][mi][ni] = (v8f){0.f,0.f,0.f,0.f,0.f,0.f,0.f,0.f};

    const int nStages = K / BLK_K;

    // Async DMA of both 128x64 bf16 B tiles into LDS buffer `buf`:
    // 2 B x 4 x b128 per thread, no VGPR staging, tracked by ASYNCcnt.
    auto loadB_async = [&](int kbase, int buf) {
#pragma unroll
        for (int b = 0; b < 2; ++b)
#pragma unroll
            for (int i = 0; i < 4; ++i) {
                int idx = tid + i * THREADS; int r = idx >> 3, c = idx & 7;
                unsigned ldsOff = (unsigned)(uintptr_t)&sB[buf][b][r][c * 8];
                const bf16_t* g = BT[b] + (size_t)(n0 + r) * ldb + kbase + c * 8;
                asm volatile("global_load_async_to_lds_b128 %0, %1, off"
                             :: "v"(ldsOff), "v"(g) : "memory");
            }
    };

    float4 aPre[8];

    // ---- stage 0 ----
    loadB_async(0, 0);
#pragma unroll
    for (int i = 0; i < 8; ++i) {
        int idx = tid + i * THREADS; int r = idx >> 4, c = idx & 15;
        aPre[i] = *reinterpret_cast<const float4*>(A + (size_t)(m0 + r) * lda + c * 4);
    }
#pragma unroll
    for (int i = 0; i < 8; ++i) {
        int idx = tid + i * THREADS; int r = idx >> 4, c = idx & 15;
        v4bf t; t[0] = (bf16_t)aPre[i].x; t[1] = (bf16_t)aPre[i].y;
                t[2] = (bf16_t)aPre[i].z; t[3] = (bf16_t)aPre[i].w;
        *reinterpret_cast<v4bf*>(&sA[r][c * 4]) = t;
    }
    asm volatile("s_wait_asynccnt 0x0" ::: "memory");
    __syncthreads();

    for (int s = 0; s < nStages; ++s) {
        const int buf = s & 1;
        const int kNext = (s + 1) * BLK_K;
        if (s + 1 < nStages) {
            // next B tiles: async DMA into the other LDS buffer, overlaps WMMA
            loadB_async(kNext, buf ^ 1);
            // next A tile: register-staged (needs f32->bf16 conversion)
#pragma unroll
            for (int i = 0; i < 8; ++i) {
                int idx = tid + i * THREADS; int r = idx >> 4, c = idx & 15;
                aPre[i] = *reinterpret_cast<const float4*>(
                    A + (size_t)(m0 + r) * lda + kNext + c * 4);
            }
        }

#pragma unroll
        for (int kk = 0; kk < BLK_K; kk += 32) {
            FragBF bF[2][2];
#pragma unroll
            for (int b = 0; b < 2; ++b)
#pragma unroll
                for (int ni = 0; ni < 2; ++ni) {
                    int rb = waveN * 32 + ni * 16 + row;
                    bF[b][ni].h[0] = *reinterpret_cast<const v8bf*>(&sB[buf][b][rb][kk + grp * 16]);
                    bF[b][ni].h[1] = *reinterpret_cast<const v8bf*>(&sB[buf][b][rb][kk + grp * 16 + 8]);
                }
#pragma unroll
            for (int mi = 0; mi < 4; ++mi) {
                int ra = waveM * 64 + mi * 16 + row;
                FragBF aF;
                aF.h[0] = *reinterpret_cast<const v8bf*>(&sA[ra][kk + grp * 16]);
                aF.h[1] = *reinterpret_cast<const v8bf*>(&sA[ra][kk + grp * 16 + 8]);
#pragma unroll
                for (int b = 0; b < 2; ++b)
#pragma unroll
                    for (int ni = 0; ni < 2; ++ni) {
                        acc[b][mi][ni] = __builtin_amdgcn_wmma_f32_16x16x32_bf16(
                            false, aF.f, false, bF[b][ni].f, (short)0, acc[b][mi][ni],
                            false, false);
                    }
            }
        }
        __syncthreads();   // all waves done reading sA (B next-buf is untouched)
        if (s + 1 < nStages) {
#pragma unroll
            for (int i = 0; i < 8; ++i) {
                int idx = tid + i * THREADS; int r = idx >> 4, c = idx & 15;
                v4bf t; t[0] = (bf16_t)aPre[i].x; t[1] = (bf16_t)aPre[i].y;
                        t[2] = (bf16_t)aPre[i].z; t[3] = (bf16_t)aPre[i].w;
                *reinterpret_cast<v4bf*>(&sA[r][c * 4]) = t;
            }
            asm volatile("s_wait_asynccnt 0x0" ::: "memory");  // this wave's B DMA done
            __syncthreads();                                   // => whole tile visible
        }
    }

    const float alphav = STORE_F32 ? alpha[0] : 0.f;
    float*  Cout[2]  = { C0,  C1 };
    bf16_t* CTout[2] = { CT0, CT1 };
#pragma unroll
    for (int b = 0; b < 2; ++b)
#pragma unroll
        for (int mi = 0; mi < 4; ++mi)
#pragma unroll
            for (int ni = 0; ni < 2; ++ni) {
                const int n  = n0 + waveN * 32 + ni * 16 + row;
                const int mB = m0 + waveM * 64 + mi * 16 + grp * 8;
                if (STORE_F32) {
                    const float bv = bias[n];
#pragma unroll
                    for (int r = 0; r < 8; ++r) {
                        float v = acc[b][mi][ni][r] + bv;
                        v = (v >= 0.f) ? v : alphav * v;
                        Cout[b][(size_t)(mB + r) * ldc + n] = v;
                    }
                } else {
                    v8bf t;
#pragma unroll
                    for (int r = 0; r < 8; ++r) t[r] = (bf16_t)acc[b][mi][ni][r];
                    *reinterpret_cast<v8bf*>(&CTout[b][(size_t)n * ldct + mB]) = t;
                }
            }
}

// ---------------------------------------------------------------------------
extern "C" void kernel_launch(void* const* d_in, const int* in_sizes, int n_in,
                              void* d_out, int out_size, void* d_ws, size_t ws_size,
                              hipStream_t stream)
{
    const int N = 8192, NIN = 512, NH = 512;

    const float* seq_pos = (const float*)d_in[0];
    const float* seq_neg = (const float*)d_in[1];
    const float* adj     = (const float*)d_in[2];
    const float* diff    = (const float*)d_in[3];
    const float* W1a = (const float*)d_in[4];
    const float* W1d = (const float*)d_in[5];
    const float* W2a = (const float*)d_in[6];
    const float* W2d = (const float*)d_in[7];
    const float* b1a = (const float*)d_in[8];
    const float* b1d = (const float*)d_in[9];
    const float* b2a = (const float*)d_in[10];
    const float* b2d = (const float*)d_in[11];
    const float* a1a = (const float*)d_in[12];
    const float* a1d = (const float*)d_in[13];
    const float* a2a = (const float*)d_in[14];
    const float* a2d = (const float*)d_in[15];

    float* out = (float*)d_out;
    const size_t OS = (size_t)N * NH;
    float* h_a1_p = out + 0 * OS;
    float* h_d1_p = out + 1 * OS;
    float* h_a2_p = out + 2 * OS;
    float* h_d2_p = out + 3 * OS;
    float* h_a1_n = out + 4 * OS;
    float* h_d1_n = out + 5 * OS;
    float* h_a2_n = out + 6 * OS;
    float* h_d2_n = out + 7 * OS;

    bf16_t* ws = (bf16_t*)d_ws;
    bf16_t* WT1a = ws;
    bf16_t* WT1d = WT1a + (size_t)NIN * NH;
    bf16_t* WT2a = WT1d + (size_t)NIN * NH;
    bf16_t* WT2d = WT2a + (size_t)NH * NH;
    bf16_t* fts0 = WT2d + (size_t)NH * NH;   // 512 x 8192 bf16, N-major
    bf16_t* fts1 = fts0 + (size_t)NH * N;
    bf16_t* fts2 = fts1 + (size_t)NH * N;
    bf16_t* fts3 = fts2 + (size_t)NH * N;

    // 0) weights -> transposed bf16
    {
        dim3 b(256), g((NIN * NH + 255) / 256);
        transpose_bf16<<<g, b, 0, stream>>>(W1a, WT1a, NIN, NH);
        transpose_bf16<<<g, b, 0, stream>>>(W1d, WT1d, NIN, NH);
        transpose_bf16<<<g, b, 0, stream>>>(W2a, WT2a, NH, NH);
        transpose_bf16<<<g, b, 0, stream>>>(W2d, WT2d, NH, NH);
    }

    dim3 blk(THREADS);
    dim3 grd(NH / BLK_N, N / BLK_M);   // all GEMMs: M=8192, N=512

    // 1) layer-1 features (fused on shared seq): fts = seq @ {W1a, W1d}
    gemm_dual_bf16_wmma<false><<<grd, blk, 0, stream>>>(
        seq_pos, NIN, WT1a, WT1d, NIN, nullptr, nullptr, 0, fts0, fts1, N,
        nullptr, nullptr, NIN);
    gemm_dual_bf16_wmma<false><<<grd, blk, 0, stream>>>(
        seq_neg, NIN, WT1a, WT1d, NIN, nullptr, nullptr, 0, fts2, fts3, N,
        nullptr, nullptr, NIN);

    // 2) layer-1 aggregation (fused pos+neg per A read): h1 = PReLU(A@fts + b)
    gemm_dual_bf16_wmma<true><<<grd, blk, 0, stream>>>(
        adj,  N, fts0, fts2, N, h_a1_p, h_a1_n, NH, nullptr, nullptr, 0, b1a, a1a, N);
    gemm_dual_bf16_wmma<true><<<grd, blk, 0, stream>>>(
        diff, N, fts1, fts3, N, h_d1_p, h_d1_n, NH, nullptr, nullptr, 0, b1d, a1d, N);

    // 3) layer-2 features (distinct A per GEMM): fts = h1 @ W2
    gemm_bf16_wmma<false><<<grd, blk, 0, stream>>>(h_a1_p, NH, WT2a, NH, nullptr, 0, fts0, N, nullptr, nullptr, NH);
    gemm_bf16_wmma<false><<<grd, blk, 0, stream>>>(h_d1_p, NH, WT2d, NH, nullptr, 0, fts1, N, nullptr, nullptr, NH);
    gemm_bf16_wmma<false><<<grd, blk, 0, stream>>>(h_a1_n, NH, WT2a, NH, nullptr, 0, fts2, N, nullptr, nullptr, NH);
    gemm_bf16_wmma<false><<<grd, blk, 0, stream>>>(h_d1_n, NH, WT2d, NH, nullptr, 0, fts3, N, nullptr, nullptr, NH);

    // 4) layer-2 aggregation (fused pos+neg per A read)
    gemm_dual_bf16_wmma<true><<<grd, blk, 0, stream>>>(
        adj,  N, fts0, fts2, N, h_a2_p, h_a2_n, NH, nullptr, nullptr, 0, b2a, a2a, N);
    gemm_dual_bf16_wmma<true><<<grd, blk, 0, stream>>>(
        diff, N, fts1, fts3, N, h_d2_p, h_d2_n, NH, nullptr, nullptr, 0, b2d, a2d, N);
}
